// EMPSNLayer_14886356648020
// MI455X (gfx1250) — matrix operand order repaired and checked by reading, not compiled, not run on hardware
//
#include <hip/hip_runtime.h>
#include <hip/hip_bf16.h>
#include <stdint.h>

#define H 128
#define NI 3

typedef __attribute__((ext_vector_type(16))) __bf16 v16bf;
typedef __attribute__((ext_vector_type(8)))  float  v8f;

union ATile { v16bf v; uint32_t u[8]; };

// f32 pair -> packed bf16 dword. Prefer the HW packed converter; else truncate.
__device__ inline uint32_t pk_bf16(float lo, float hi) {
#if __has_builtin(__builtin_amdgcn_cvt_pk_bf16_f32)
  typedef __bf16 v2bf __attribute__((ext_vector_type(2)));
  union { v2bf v; uint32_t u; } c;
  c.v = __builtin_amdgcn_cvt_pk_bf16_f32(lo, hi);
  return c.u;
#else
  return (__float_as_uint(lo) >> 16) | (__float_as_uint(hi) & 0xFFFF0000u);
#endif
}

// load 8 consecutive f32, pack to 4 dwords of bf16
__device__ inline void load8f_pk(const float* __restrict__ p, uint32_t* d) {
  float4 a = *reinterpret_cast<const float4*>(p);
  float4 b = *reinterpret_cast<const float4*>(p + 4);
  d[0] = pk_bf16(a.x, a.y);
  d[1] = pk_bf16(a.z, a.w);
  d[2] = pk_bf16(b.x, b.y);
  d[3] = pk_bf16(b.z, b.w);
}

// load 8 consecutive bf16 (16 bytes)
__device__ inline void load8bf(const uint16_t* __restrict__ p, uint32_t* d) {
  uint4 t = *reinterpret_cast<const uint4*>(p);
  d[0] = t.x; d[1] = t.y; d[2] = t.z; d[3] = t.w;
}

__device__ inline float fast_sigmoid(float x) {
  return 1.0f / (1.0f + __expf(-x));
}

__device__ inline v8f wmma_bf16(const ATile& a, const ATile& b, v8f c) {
  return __builtin_amdgcn_wmma_f32_16x16x32_bf16(false, a.v, false, b.v,
                                                 (short)0, c, false, false);
}

// -----------------------------------------------------------------------------
// Elementwise f32 -> bf16 (8 elements/thread, vectorized b128 in/out)
// -----------------------------------------------------------------------------
__global__ void conv_x_kernel(const float* __restrict__ x,
                              uint16_t* __restrict__ xb, long n8) {
  long i = (long)blockIdx.x * blockDim.x + threadIdx.x;
  if (i >= n8) return;
  uint32_t d[4];
  load8f_pk(x + i * 8, d);
  *reinterpret_cast<uint4*>(xb + i * 8) = make_uint4(d[0], d[1], d[2], d[3]);
}

// -----------------------------------------------------------------------------
// Weight transpose + f32->bf16: WT[n*KW + k] = bf16(W[k*H + n]) for k < K
// -----------------------------------------------------------------------------
__global__ void conv_wt_kernel(const float* __restrict__ W,
                               uint16_t* __restrict__ WT, int K, int KW) {
  int idx = blockIdx.x * blockDim.x + threadIdx.x;
  if (idx >= H * KW) return;
  int n = idx / KW, k = idx % KW;
  float v = (k < K) ? W[(size_t)k * H + n] : 0.0f;
  WT[idx] = (uint16_t)(pk_bf16(v, 0.0f) & 0xFFFFu);
}

// -----------------------------------------------------------------------------
// Message passing: block = 64 edges (4 M-tiles) x 128 cols, 4 waves.
// Per K-chunk: one clause of 12 b128 loads (4 B + 8 A), then 8 WMMAs.
// Row pointers hoisted out of the K-loop (no address VALU between WMMAs).
// -----------------------------------------------------------------------------
__global__ __launch_bounds__(128) void mp_kernel(
    const uint16_t* __restrict__ xs, const uint16_t* __restrict__ xr,  // bf16 [N][H]
    const int* __restrict__ idx_s, const int* __restrict__ idx_r,
    const float* __restrict__ inv,
    const uint16_t* __restrict__ WmT,  // [H][256] bf16 (transposed, first 2H rows)
    const float* __restrict__ Wm,      // [2H+NI][H] f32 (for the inv rank-3 rows)
    const float* __restrict__ bm, const float* __restrict__ Wi,
    const float* __restrict__ bi,
    float* __restrict__ mOut, int E) {
  __shared__ int   s_sidx[64];
  __shared__ int   s_ridx[64];
  __shared__ float s_inv[64][4];
  __shared__ float s_dot[64];
  __shared__ float s_w[64];

  const int tid  = threadIdx.x;
  const int wave = tid >> 5;
  const int lane = tid & 31;
  const int half = lane >> 4;
  const int l16  = lane & 15;
  const long ebase = (long)blockIdx.x * 64;

  if (tid < 64) {
    long e = ebase + tid;
    bool ok = (e < (long)E);
    long ec = ok ? e : 0;
    s_sidx[tid]  = idx_s[ec];
    s_ridx[tid]  = ok ? idx_r[ec] : -1;
    s_inv[tid][0] = ok ? inv[ec * NI + 0] : 0.0f;
    s_inv[tid][1] = ok ? inv[ec * NI + 1] : 0.0f;
    s_inv[tid][2] = ok ? inv[ec * NI + 2] : 0.0f;
    s_dot[tid] = 0.0f;
  }
  __syncthreads();

  const int col0 = wave * 32 + l16;  // N for tile 0
  const int col1 = col0 + 16;        // N for tile 1
  const uint16_t* bp0 = WmT + (size_t)col0 * 256;
  const uint16_t* bp1 = WmT + (size_t)col1 * 256;

  // hoisted gathered-row pointers (A layout: this lane reads row l16 of tile mt)
  const uint16_t* sptr[4];
  const uint16_t* rptr[4];
#pragma unroll
  for (int mt = 0; mt < 4; ++mt) {
    int sr = s_sidx[mt * 16 + l16];
    int rr = s_ridx[mt * 16 + l16]; if (rr < 0) rr = 0;
    sptr[mt] = xs + (size_t)sr * H + half * 8;
    rptr[mt] = xr + (size_t)rr * H + half * 8;
  }

  v8f acc[4][2];
#pragma unroll
  for (int mt = 0; mt < 4; ++mt) { acc[mt][0] = (v8f){}; acc[mt][1] = (v8f){}; }

#pragma unroll
  for (int kc = 0; kc < 8; ++kc) {
    const int koff = (kc & 3) * 32;       // element offset within source row
    const int kg = kc * 32 + half * 8;    // element offset within WmT row
    ATile b0, b1, a[4];
    load8bf(bp0 + kg, b0.u);
    load8bf(bp0 + kg + 16, b0.u + 4);
    load8bf(bp1 + kg, b1.u);
    load8bf(bp1 + kg + 16, b1.u + 4);
#pragma unroll
    for (int mt = 0; mt < 4; ++mt) {
      const uint16_t* ap = ((kc < 4) ? sptr[mt] : rptr[mt]) + koff;
      load8bf(ap, a[mt].u);        // K {0..7 | 8..15} of chunk
      load8bf(ap + 16, a[mt].u + 4);  // K {16..23 | 24..31}
    }
#pragma unroll
    for (int mt = 0; mt < 4; ++mt) {
      acc[mt][0] = wmma_bf16(a[mt], b0, acc[mt][0]);
      acc[mt][1] = wmma_bf16(a[mt], b1, acc[mt][1]);
    }
  }

  // rank-3 inv contribution + bias + SiLU + partial gate dot (in-place in acc)
  float w3_0[3], w3_1[3];
#pragma unroll
  for (int i = 0; i < 3; ++i) {
    w3_0[i] = Wm[(size_t)(2 * H + i) * H + col0];
    w3_1[i] = Wm[(size_t)(2 * H + i) * H + col1];
  }
  const float bm0 = bm[col0], bm1 = bm[col1];
  const float wi0 = Wi[col0], wi1 = Wi[col1];
#pragma unroll
  for (int mt = 0; mt < 4; ++mt) {
#pragma unroll
    for (int v = 0; v < 8; ++v) {
      const int M = mt * 16 + v + half * 8;  // edge row within block (C/D layout)
      const float i0 = s_inv[M][0], i1 = s_inv[M][1], i2 = s_inv[M][2];
      float m0 = acc[mt][0][v] + bm0 + i0 * w3_0[0] + i1 * w3_0[1] + i2 * w3_0[2];
      float m1 = acc[mt][1][v] + bm1 + i0 * w3_1[0] + i1 * w3_1[1] + i2 * w3_1[2];
      float s0 = m0 * fast_sigmoid(m0);
      float s1 = m1 * fast_sigmoid(m1);
      acc[mt][0][v] = s0;
      acc[mt][1][v] = s1;
      atomicAdd(&s_dot[M], s0 * wi0 + s1 * wi1);  // ds_add_f32
    }
  }
  __syncthreads();
  if (tid < 64) s_w[tid] = fast_sigmoid(s_dot[tid] + bi[0]);
  __syncthreads();

#pragma unroll
  for (int mt = 0; mt < 4; ++mt) {
#pragma unroll
    for (int v = 0; v < 8; ++v) {
      const int M = mt * 16 + v + half * 8;
      const int r = s_ridx[M];
      if (r >= 0) {
        const float g = s_w[M];
        unsafeAtomicAdd(&mOut[(size_t)r * H + col0], acc[mt][0][v] * g);
        unsafeAtomicAdd(&mOut[(size_t)r * H + col1], acc[mt][1][v] * g);
      }
    }
  }
}

// -----------------------------------------------------------------------------
// Update: out = x + concat(x, mA[, mB]) @ Wu + bu.  NC = K/32 (8 or 12).
// 64 rows per block (4 M-tiles), B reused across M-tiles, pointers hoisted.
// -----------------------------------------------------------------------------
template <int NC>
__global__ __launch_bounds__(128) void update_kernel(
    const float* __restrict__ x, const uint16_t* __restrict__ xb,
    const float* __restrict__ mA, const float* __restrict__ mB,
    const uint16_t* __restrict__ WuT,  // [H][NC*32] bf16 transposed
    const float* __restrict__ bu,
    float* __restrict__ out, int N) {
  const int tid  = threadIdx.x;
  const int wave = tid >> 5;
  const int lane = tid & 31;
  const int half = lane >> 4;
  const int l16  = lane & 15;
  const long rbase = (long)blockIdx.x * 64;

  const uint16_t* xbp[4];
  const float* map[4];
  const float* mbp[4];
#pragma unroll
  for (int mt = 0; mt < 4; ++mt) {
    long lr = rbase + mt * 16 + l16;
    if (lr >= N) lr = N - 1;
    xbp[mt] = xb + (size_t)lr * H + half * 8;
    map[mt] = mA + (size_t)lr * H + half * 8;
    mbp[mt] = mB + (size_t)lr * H + half * 8;
  }

  const int KW = NC * 32;
  const int col0 = wave * 32 + l16;
  const int col1 = col0 + 16;
  const uint16_t* bp0 = WuT + (size_t)col0 * KW;
  const uint16_t* bp1 = WuT + (size_t)col1 * KW;

  v8f acc[4][2];
#pragma unroll
  for (int mt = 0; mt < 4; ++mt) { acc[mt][0] = (v8f){}; acc[mt][1] = (v8f){}; }

#pragma unroll
  for (int kc = 0; kc < NC; ++kc) {
    const int koff = (kc & 3) * 32;
    const int kg = kc * 32 + half * 8;
    ATile b0, b1;
    load8bf(bp0 + kg, b0.u);
    load8bf(bp0 + kg + 16, b0.u + 4);
    load8bf(bp1 + kg, b1.u);
    load8bf(bp1 + kg + 16, b1.u + 4);
#pragma unroll
    for (int mt = 0; mt < 4; ++mt) {
      ATile a;
      if (kc < 4) {
        const uint16_t* ap = xbp[mt] + koff;
        load8bf(ap, a.u);
        load8bf(ap + 16, a.u + 4);
      } else {
        const float* ap = ((kc < 8) ? map[mt] : mbp[mt]) + koff;
        load8f_pk(ap, a.u);
        load8f_pk(ap + 16, a.u + 4);
      }
      acc[mt][0] = wmma_bf16(a, b0, acc[mt][0]);
      acc[mt][1] = wmma_bf16(a, b1, acc[mt][1]);
    }
  }

  const float bu0 = bu[col0], bu1 = bu[col1];
#pragma unroll
  for (int mt = 0; mt < 4; ++mt) {
#pragma unroll
    for (int v = 0; v < 8; ++v) {
      const int M = mt * 16 + v + half * 8;
      const long r = rbase + M;
      if (r < N) {
        out[r * H + col0] = x[r * H + col0] + acc[mt][0][v] + bu0;
        out[r * H + col1] = x[r * H + col1] + acc[mt][1][v] + bu1;
      }
    }
  }
}

// -----------------------------------------------------------------------------
extern "C" void kernel_launch(void* const* d_in, const int* in_sizes, int n_in,
                              void* d_out, int out_size, void* d_ws, size_t ws_size,
                              hipStream_t stream) {
  const float* x0 = (const float*)d_in[0];
  const float* x1 = (const float*)d_in[1];
  const float* x2 = (const float*)d_in[2];
  const int* adj00 = (const int*)d_in[3];
  const int* adj01 = (const int*)d_in[4];
  const int* adj11 = (const int*)d_in[5];
  const int* adj12 = (const int*)d_in[6];
  const int* adj22 = (const int*)d_in[7];
  const float* inv00 = (const float*)d_in[8];
  const float* inv01 = (const float*)d_in[9];
  const float* inv11 = (const float*)d_in[10];
  const float* inv12 = (const float*)d_in[11];
  const float* inv22 = (const float*)d_in[12];
  const float* Wm[5] = {(const float*)d_in[13], (const float*)d_in[17],
                        (const float*)d_in[21], (const float*)d_in[25],
                        (const float*)d_in[29]};
  const float* bm[5] = {(const float*)d_in[14], (const float*)d_in[18],
                        (const float*)d_in[22], (const float*)d_in[26],
                        (const float*)d_in[30]};
  const float* Wi[5] = {(const float*)d_in[15], (const float*)d_in[19],
                        (const float*)d_in[23], (const float*)d_in[27],
                        (const float*)d_in[31]};
  const float* bi[5] = {(const float*)d_in[16], (const float*)d_in[20],
                        (const float*)d_in[24], (const float*)d_in[28],
                        (const float*)d_in[32]};
  const float* Wu0 = (const float*)d_in[33]; const float* bu0 = (const float*)d_in[34];
  const float* Wu1 = (const float*)d_in[35]; const float* bu1 = (const float*)d_in[36];
  const float* Wu2 = (const float*)d_in[37]; const float* bu2 = (const float*)d_in[38];

  const int N0 = in_sizes[0] / H;
  const int N1 = in_sizes[1] / H;
  const int N2 = in_sizes[2] / H;
  const int E00 = in_sizes[3] / 2, E01 = in_sizes[4] / 2, E11 = in_sizes[5] / 2;
  const int E12 = in_sizes[6] / 2, E22 = in_sizes[7] / 2;

  // workspace layout
  char* ws = (char*)d_ws;
  float* m00 = (float*)ws; ws += (size_t)N0 * H * 4;
  float* m01 = (float*)ws; ws += (size_t)N1 * H * 4;
  float* m11 = (float*)ws; ws += (size_t)N1 * H * 4;
  float* m12 = (float*)ws; ws += (size_t)N2 * H * 4;
  float* m22 = (float*)ws; ws += (size_t)N2 * H * 4;
  uint16_t* xb0 = (uint16_t*)ws; ws += (size_t)N0 * H * 2;
  uint16_t* xb1 = (uint16_t*)ws; ws += (size_t)N1 * H * 2;
  uint16_t* xb2 = (uint16_t*)ws; ws += (size_t)N2 * H * 2;
  uint16_t* WmT[5];
  for (int t = 0; t < 5; ++t) { WmT[t] = (uint16_t*)ws; ws += (size_t)H * 256 * 2; }
  uint16_t* WuT0 = (uint16_t*)ws; ws += (size_t)H * 256 * 2;
  uint16_t* WuT1 = (uint16_t*)ws; ws += (size_t)H * 384 * 2;
  uint16_t* WuT2 = (uint16_t*)ws; ws += (size_t)H * 384 * 2;

  // zero all message buffers (contiguous at start of ws)
  size_t mbytes = ((size_t)N0 + 2 * (size_t)N1 + 2 * (size_t)N2) * H * 4;
  hipMemsetAsync(d_ws, 0, mbytes, stream);

  // feature conversion to bf16
  {
    long n8_0 = (long)N0 * H / 8, n8_1 = (long)N1 * H / 8, n8_2 = (long)N2 * H / 8;
    conv_x_kernel<<<(int)((n8_0 + 255) / 256), 256, 0, stream>>>(x0, xb0, n8_0);
    conv_x_kernel<<<(int)((n8_1 + 255) / 256), 256, 0, stream>>>(x1, xb1, n8_1);
    conv_x_kernel<<<(int)((n8_2 + 255) / 256), 256, 0, stream>>>(x2, xb2, n8_2);
  }

  // transpose + convert weights to bf16
  {
    dim3 blk(256);
    dim3 g256((H * 256 + 255) / 256), g384((H * 384 + 255) / 256);
    for (int t = 0; t < 5; ++t)
      conv_wt_kernel<<<g256, blk, 0, stream>>>(Wm[t], WmT[t], 256, 256);
    conv_wt_kernel<<<g256, blk, 0, stream>>>(Wu0, WuT0, 256, 256);
    conv_wt_kernel<<<g384, blk, 0, stream>>>(Wu1, WuT1, 384, 384);
    conv_wt_kernel<<<g384, blk, 0, stream>>>(Wu2, WuT2, 384, 384);
  }

  // message passing (5 adjacency types)
  mp_kernel<<<(E00 + 63) / 64, 128, 0, stream>>>(xb0, xb0, adj00, adj00 + E00,
      inv00, WmT[0], Wm[0], bm[0], Wi[0], bi[0], m00, E00);
  mp_kernel<<<(E01 + 63) / 64, 128, 0, stream>>>(xb0, xb1, adj01, adj01 + E01,
      inv01, WmT[1], Wm[1], bm[1], Wi[1], bi[1], m01, E01);
  mp_kernel<<<(E11 + 63) / 64, 128, 0, stream>>>(xb1, xb1, adj11, adj11 + E11,
      inv11, WmT[2], Wm[2], bm[2], Wi[2], bi[2], m11, E11);
  mp_kernel<<<(E12 + 63) / 64, 128, 0, stream>>>(xb1, xb2, adj12, adj12 + E12,
      inv12, WmT[3], Wm[3], bm[3], Wi[3], bi[3], m12, E12);
  mp_kernel<<<(E22 + 63) / 64, 128, 0, stream>>>(xb2, xb2, adj22, adj22 + E22,
      inv22, WmT[4], Wm[4], bm[4], Wi[4], bi[4], m22, E22);

  // node updates with residual
  float* out = (float*)d_out;
  float* out0 = out;
  float* out1 = out + (size_t)N0 * H;
  float* out2 = out + ((size_t)N0 + N1) * H;
  update_kernel<8><<<(N0 + 63) / 64, 128, 0, stream>>>(x0, xb0, m00, m00, WuT0, bu0, out0, N0);
  update_kernel<12><<<(N1 + 63) / 64, 128, 0, stream>>>(x1, xb1, m01, m11, WuT1, bu1, out1, N1);
  update_kernel<12><<<(N2 + 63) / 64, 128, 0, stream>>>(x2, xb2, m12, m22, WuT2, bu2, out2, N2);
}
